// Model_73014444032001
// MI455X (gfx1250) — matrix operand order repaired and checked by reading, not compiled, not run on hardware
//
#include <hip/hip_runtime.h>
#include <hip/hip_bf16.h>
#include <stdint.h>

// ---------------------------------------------------------------------------
// Flash attention forward for B=2, H=16, S=2048, D=64 (fp32 in/out).
// bf16 WMMA (v_wmma_f32_16x16x32_bf16) with fp32 accumulation, wave32.
// - KTILE=128 KV tiles: per-tile softmax fixed costs (permlane max butterflies,
//   alpha, O rescale) amortize over 2x the WMMA work; 16 barriers total.
// - Double-buffered LDS K/V tiles, register-pipelined global loads.
// - Softmax in log2 domain (v_exp_f32 is base-2): 0.125*log2(e) folded into Q.
// - Row sums by WMMA against a constant all-ones B fragment.
// - Row max via v_permlane16_b32 butterflies (VALU pipe, not LDS pipe).
// ---------------------------------------------------------------------------

typedef __attribute__((ext_vector_type(16))) __bf16 v16bf;
typedef __attribute__((ext_vector_type(2)))  __bf16 v2bf;
typedef __attribute__((ext_vector_type(8)))  float  v8f;

union Frag16 {
    v16bf v;
    uint32_t u[8];
};

__device__ __forceinline__ uint16_t f32_to_bf16(float x) {
    __bf16 h = (__bf16)x;
    return __builtin_bit_cast(uint16_t, h);
}

__device__ __forceinline__ uint32_t pack2_bf16(float a, float b) {
    v2bf p;
    p.x = (__bf16)a;
    p.y = (__bf16)b;
    return __builtin_bit_cast(uint32_t, p);
}

__device__ __forceinline__ v8f wmma_bf16(v16bf a, v16bf b, v8f c) {
    // args: (neg_a, A, neg_b, B, c_mod, C, reuse_a, reuse_b)
    return __builtin_amdgcn_wmma_f32_16x16x32_bf16(
        false, a, false, b, (short)0, c, false, false);
}

__device__ __forceinline__ float fast_exp2(float x) {
#if __has_builtin(__builtin_amdgcn_exp2f)
    return __builtin_amdgcn_exp2f(x);        // raw v_exp_f32
#else
    return exp2f(x);
#endif
}

__device__ __forceinline__ float xor16_max(float x, uint32_t sel_lo, uint32_t sel_hi) {
#if __has_builtin(__builtin_amdgcn_permlane16)
    int xi = __builtin_bit_cast(int, x);
    int pi = __builtin_amdgcn_permlane16(xi, xi, sel_lo, sel_hi, false, false);
    return fmaxf(x, __builtin_bit_cast(float, pi));
#else
    return fmaxf(x, __shfl_xor(x, 0, 16));
#endif
}

__device__ __forceinline__ float rowmax16(float x) {
#if __has_builtin(__builtin_amdgcn_permlane16)
    x = xor16_max(x, 0x67452301u, 0xEFCDAB89u);   // xor 1
    x = xor16_max(x, 0x54761032u, 0xDCFE98BAu);   // xor 2
    x = xor16_max(x, 0x32107654u, 0xBA98FEDCu);   // xor 4
    x = xor16_max(x, 0xFEDCBA98u, 0x76543210u);   // xor 8
#else
    for (int msk = 8; msk >= 1; msk >>= 1)
        x = fmaxf(x, __shfl_xor(x, msk, 16));
#endif
    return x;
}

#define ATTN_B 2
#define ATTN_H 16
#define ATTN_S 2048
#define ATTN_D 64
#define QTILE  128     // query rows per block (16 per wave x 8 waves)
#define KTILE  128     // kv rows per iteration
#define NCHUNK (KTILE / 16)   // 8 key chunks of 16

// 1/sqrt(64) * log2(e): softmax computed with exp2.
#define QK_SCALE_LOG2E 0.1803368801111204f

__global__ __launch_bounds__(256)
void fa_fwd_bf16_wmma(const float* __restrict__ Q,
                      const float* __restrict__ K,
                      const float* __restrict__ V,
                      float* __restrict__ O) {
    // Double-buffered staging: K row-major [kv][d], V transposed [d][kv] (bf16).
    __shared__ __align__(16) uint16_t sK [2][KTILE * ATTN_D];   // 2 x 16 KB
    __shared__ __align__(16) uint16_t sVt[2][ATTN_D * KTILE];   // 2 x 16 KB
    __shared__ __align__(16) uint16_t sP [8 * 16 * KTILE];      // 32 KB per-wave P

    const int t    = threadIdx.x;          // 0..255
    const int wave = t >> 5;               // 0..7
    const int lane = t & 31;
    const int ln   = lane & 15;            // lane within 16-group
    const int hi   = lane >> 4;            // which half of the wave

    const int qblk = blockIdx.x;           // 0..S/QTILE-1
    const int bh   = blockIdx.y;           // 0..B*H-1

    const float* Qbh = Q + (size_t)bh * ATTN_S * ATTN_D;
    const float* Kbh = K + (size_t)bh * ATTN_S * ATTN_D;
    const float* Vbh = V + (size_t)bh * ATTN_S * ATTN_D;
    float*       Obh = O + (size_t)bh * ATTN_S * ATTN_D;

    const int q0 = qblk * QTILE + wave * 16;   // this wave's first query row

    // ---- Load Q tile (16x64) as two 16x32 bf16 A-fragments; fold in
    // 1/8 * log2(e) so scores come out in the log2 domain.
    // A operand, contiguous k order used here (B matches): lane L -> M=ln; VGPR v holds
    // K = 2*v + hi*16 (and K+1 in the high half of the dword).
    Frag16 qa[2];
    {
        const float* qp = Qbh + (size_t)(q0 + ln) * ATTN_D;
#pragma unroll
        for (int half = 0; half < 2; ++half) {
#pragma unroll
            for (int v = 0; v < 8; ++v) {
                const int k = half * 32 + 2 * v + hi * 16;
                qa[half].u[v] = pack2_bf16(qp[k] * QK_SCALE_LOG2E,
                                           qp[k + 1] * QK_SCALE_LOG2E);
            }
        }
    }

    // Constant all-ones 32x16 bf16 B fragment for WMMA row sums.
    Frag16 ones;
#pragma unroll
    for (int v = 0; v < 8; ++v) ones.u[v] = 0x3F803F80u;   // (1.0bf16, 1.0bf16)

    // ---- Online-softmax state. C layout: VGPR r, lane L: row=r+8*hi, col=ln.
    v8f   o[4];
    float m[8], l[8];
#pragma unroll
    for (int c = 0; c < 4; ++c) o[c] = (v8f)0.0f;
#pragma unroll
    for (int r = 0; r < 8; ++r) { m[r] = -1.0e30f; l[r] = 0.0f; }

    // Cooperative-staging coordinates: 2 threads per kv row, 32 d each.
    const int srow  = t >> 1;              // 0..127
    const int dhalf = (t & 1) << 5;        // 0 or 32

    uint16_t* Pw = &sP[wave * 16 * KTILE];

    // ---- Software pipeline: tile kt+KTILE loads overlap tile kt compute.
    float4 kr[8], vr[8];
    {
        const float4* kg = reinterpret_cast<const float4*>(
            Kbh + (size_t)srow * ATTN_D + dhalf);
        const float4* vg = reinterpret_cast<const float4*>(
            Vbh + (size_t)srow * ATTN_D + dhalf);
#pragma unroll
        for (int i = 0; i < 8; ++i) { kr[i] = kg[i]; vr[i] = vg[i]; }
    }

    for (int kt = 0; kt < ATTN_S; kt += KTILE) {
        const int buf = (kt >> 7) & 1;
        uint16_t* sKb = sK[buf];
        uint16_t* sVb = sVt[buf];

        // ---- Commit staged registers to this tile's LDS buffers.
        {
            uint4* kd = reinterpret_cast<uint4*>(&sKb[srow * ATTN_D + dhalf]);
#pragma unroll
            for (int i = 0; i < 4; ++i) {
                uint4 w = { pack2_bf16(kr[2*i].x, kr[2*i].y),
                            pack2_bf16(kr[2*i].z, kr[2*i].w),
                            pack2_bf16(kr[2*i+1].x, kr[2*i+1].y),
                            pack2_bf16(kr[2*i+1].z, kr[2*i+1].w) };
                kd[i] = w;
            }
            float vv[32];
#pragma unroll
            for (int i = 0; i < 8; ++i) {
                vv[4*i]   = vr[i].x; vv[4*i+1] = vr[i].y;
                vv[4*i+2] = vr[i].z; vv[4*i+3] = vr[i].w;
            }
#pragma unroll
            for (int i = 0; i < 32; ++i)
                sVb[(dhalf + i) * KTILE + srow] = f32_to_bf16(vv[i]);
        }
        __syncthreads();   // single hard barrier per tile (double-buffered)

        // ---- Kick off next tile's global loads; they complete during compute.
        if (kt + KTILE < ATTN_S) {
            const float4* kg = reinterpret_cast<const float4*>(
                Kbh + (size_t)(kt + KTILE + srow) * ATTN_D + dhalf);
            const float4* vg = reinterpret_cast<const float4*>(
                Vbh + (size_t)(kt + KTILE + srow) * ATTN_D + dhalf);
#pragma unroll
            for (int i = 0; i < 8; ++i) { kr[i] = kg[i]; vr[i] = vg[i]; }
        }

        // ---- S = Q * K^T  (log2-domain scores: scale folded into Q).
        // B operand, contiguous k order as for A: VGPR v, lane L: N=ln, K=2v+16*hi (+1 hi half).
        v8f sc[NCHUNK];
#pragma unroll
        for (int n = 0; n < NCHUNK; ++n) {
            Frag16 kb0, kb1;
            const int key = n * 16 + ln;
#pragma unroll
            for (int v = 0; v < 8; ++v) {
                const int kk = 2 * v + 16 * hi;
                kb0.u[v] = *reinterpret_cast<const uint32_t*>(&sKb[key * ATTN_D + kk]);
                kb1.u[v] = *reinterpret_cast<const uint32_t*>(&sKb[key * ATTN_D + 32 + kk]);
            }
            v8f c = (v8f)0.0f;
            c = wmma_bf16(qa[0].v, kb0.v, c);
            c = wmma_bf16(qa[1].v, kb1.v, c);
            sc[n] = c;
        }

        // ---- Online softmax (log2 domain). Row max via permlane butterflies.
        float mnew[8], alpha[8];
#pragma unroll
        for (int r = 0; r < 8; ++r) {
            float m0 = fmaxf(fmaxf(sc[0][r], sc[1][r]), fmaxf(sc[2][r], sc[3][r]));
            float m1 = fmaxf(fmaxf(sc[4][r], sc[5][r]), fmaxf(sc[6][r], sc[7][r]));
            float mt = rowmax16(fmaxf(m0, m1));
            mnew[r]  = fmaxf(m[r], mt);
            alpha[r] = fast_exp2(m[r] - mnew[r]);
            m[r]     = mnew[r];
        }

        // ---- P = exp2(S - m): bf16 into wave-private LDS scratch (C-layout addr).
#pragma unroll
        for (int n = 0; n < NCHUNK; ++n) {
#pragma unroll
            for (int r = 0; r < 8; ++r) {
                const float p = fast_exp2(sc[n][r] - mnew[r]);
                Pw[(r + 8 * hi) * KTILE + n * 16 + ln] = f32_to_bf16(p);
            }
        }
        // Rescale output accumulators while the P stores drain.
#pragma unroll
        for (int c = 0; c < 4; ++c)
#pragma unroll
            for (int r = 0; r < 8; ++r)
                o[c][r] *= alpha[r];

        // P is wave-private and per-wave LDS ops are in-order (DScnt ordering):
        // only a compiler fence is needed, no workgroup barrier.
        asm volatile("" ::: "memory");
        __builtin_amdgcn_wave_barrier();

        // ---- Reload P in A layout (four 16x32 bf16 fragments).
        Frag16 pa[4];
#pragma unroll
        for (int half = 0; half < 4; ++half) {
#pragma unroll
            for (int v = 0; v < 8; ++v) {
                const int k = half * 32 + 2 * v + hi * 16;
                pa[half].u[v] = *reinterpret_cast<const uint32_t*>(&Pw[ln * KTILE + k]);
            }
        }

        // ---- Row sums by WMMA against all-ones B: result broadcast in C layout.
        v8f sum = (v8f)0.0f;
        sum = wmma_bf16(pa[0].v, ones.v, sum);
        sum = wmma_bf16(pa[1].v, ones.v, sum);
        sum = wmma_bf16(pa[2].v, ones.v, sum);
        sum = wmma_bf16(pa[3].v, ones.v, sum);
#pragma unroll
        for (int r = 0; r < 8; ++r)
            l[r] = l[r] * alpha[r] + sum[r];

        // ---- O += P * V  (B fragments from transposed V: pairs contiguous in kv).
#pragma unroll
        for (int c = 0; c < 4; ++c) {
            const int drow = c * 16 + ln;
#pragma unroll
            for (int h = 0; h < 4; ++h) {
                Frag16 vb;
#pragma unroll
                for (int v = 0; v < 8; ++v) {
                    const int kk = h * 32 + 2 * v + 16 * hi;
                    vb.u[v] = *reinterpret_cast<const uint32_t*>(&sVb[drow * KTILE + kk]);
                }
                o[c] = wmma_bf16(pa[h].v, vb.v, o[c]);
            }
        }
    }

    // ---- Epilogue: O / l, fp32 store.
#pragma unroll
    for (int c = 0; c < 4; ++c) {
#pragma unroll
        for (int r = 0; r < 8; ++r) {
            const int row = q0 + r + 8 * hi;
            const int col = c * 16 + ln;
            Obh[(size_t)row * ATTN_D + col] = o[c][r] * __frcp_rn(l[r]);
        }
    }
}

extern "C" void kernel_launch(void* const* d_in, const int* in_sizes, int n_in,
                              void* d_out, int out_size, void* d_ws, size_t ws_size,
                              hipStream_t stream) {
    (void)in_sizes; (void)n_in; (void)out_size; (void)d_ws; (void)ws_size;
    const float* q = (const float*)d_in[0];
    const float* k = (const float*)d_in[1];
    const float* v = (const float*)d_in[2];
    float* out = (float*)d_out;

    dim3 grid(ATTN_S / QTILE, ATTN_B * ATTN_H);   // (16, 32)
    dim3 block(256);
    fa_fwd_bf16_wmma<<<grid, block, 0, stream>>>(q, k, v, out);
}
